// TemperNet_42099269435653
// MI455X (gfx1250) — matrix operand order repaired and checked by reading, not compiled, not run on hardware
//
#include <hip/hip_runtime.h>
#include <hip/hip_bf16.h>

// ---- problem constants (match reference) ----
#define BB   8192
#define DD   1024
#define HH   2048
#define OO   1024
#define TT   4
#define NOPS 3

typedef __attribute__((ext_vector_type(16))) __bf16 v16bf;
typedef __attribute__((ext_vector_type(8)))  float  v8f;

union Frag32B { v16bf v; float4 q[2]; };

#define KTILE 32
#define LDS_P 40   // 32 + 8 pad; 80B rows keep 16B chunks aligned, bank-skewed

// Async DMA global -> LDS, 16B, tracked by ASYNCcnt (no VGPR round-trip).
// Low 32 bits of a generic shared pointer are the LDS byte address
#define ASYNC_LOAD_B128(ldsptr, gptr)                                   \
    asm volatile("global_load_async_to_lds_b128 %0, %1, off"            \
                 :: "v"((unsigned)(uintptr_t)(ldsptr)), "v"(gptr)       \
                 : "memory")

#define WAIT_ASYNC(n) asm volatile("s_wait_asynccnt " #n ::: "memory")

// ---------------------------------------------------------------------------
// top-2 routing over 3 logits per temper (softmax is monotonic -> argmax on
// raw logits; strict '>' keeps lowest index on ties, matching jax.lax.top_k)
// ---------------------------------------------------------------------------
__global__ void route_top2(const float* __restrict__ logits, int* __restrict__ idx) {
    int t = threadIdx.x;
    if (t < TT) {
        const float* l = logits + t * NOPS;
        int i0 = 0;
        for (int i = 1; i < NOPS; ++i) if (l[i] > l[i0]) i0 = i;
        int i1 = -1;
        for (int i = 0; i < NOPS; ++i)
            if (i != i0 && (i1 < 0 || l[i] > l[i1])) i1 = i;
        idx[2 * t + 0] = i0;
        idx[2 * t + 1] = i1;
    }
}

// ---------------------------------------------------------------------------
// fp32 -> bf16 conversion, 8 elements per thread
// ---------------------------------------------------------------------------
__global__ void cvt_f32_bf16(const float* __restrict__ in, __bf16* __restrict__ out, long n) {
    long i = ((long)blockIdx.x * blockDim.x + threadIdx.x) * 8;
    if (i + 8 <= n) {
        float4 a = *(const float4*)(in + i);
        float4 b = *(const float4*)(in + i + 4);
        union { __bf16 h[8]; float4 f; } o;
        o.h[0] = (__bf16)a.x; o.h[1] = (__bf16)a.y; o.h[2] = (__bf16)a.z; o.h[3] = (__bf16)a.w;
        o.h[4] = (__bf16)b.x; o.h[5] = (__bf16)b.y; o.h[6] = (__bf16)b.z; o.h[7] = (__bf16)b.w;
        *(float4*)(out + i) = o.f;
    } else {
        for (long j = i; j < n; ++j) out[j] = (__bf16)in[j];
    }
}

// ---------------------------------------------------------------------------
// one pipeline stage: 8 WMMAs on a 128x32 LDS tile pair (32x64 per wave)
// ---------------------------------------------------------------------------
__device__ __forceinline__ void compute_tile(
    const __bf16* __restrict__ a_s, const __bf16* __restrict__ b_s,
    v8f acc[2][4], int wm, int wn, int laneLo, int laneHi)
{
    // A fragments: 16-bit A 16x32 layout — lane L holds M=L%16;
    // K runs [base, base+8) and [base+16, base+24), base = L<16 ? 0 : 8
    Frag32B afr[2];
#pragma unroll
    for (int mi = 0; mi < 2; ++mi) {
        int mr = wm * 32 + mi * 16 + laneLo;
        int kr = laneHi << 3;
        afr[mi].q[0] = *(const float4*)&a_s[mr * LDS_P + kr];
        afr[mi].q[1] = *(const float4*)&a_s[mr * LDS_P + kr + 16];
    }
    // B fragments: lane holds col N=L%16; K runs [base, base+8) and [base+16, base+24), base = L<16 ? 0 : 8
    // (W is [N,K] row-major, so rows of sB are B columns)
#pragma unroll
    for (int ni = 0; ni < 4; ++ni) {
        int nr = wn * 64 + ni * 16 + laneLo;
        int kb = laneHi << 3;
        Frag32B bfr;
        bfr.q[0] = *(const float4*)&b_s[nr * LDS_P + kb];
        bfr.q[1] = *(const float4*)&b_s[nr * LDS_P + kb + 16];
#pragma unroll
        for (int mi = 0; mi < 2; ++mi)
            acc[mi][ni] = __builtin_amdgcn_wmma_f32_16x16x32_bf16(
                false, afr[mi].v, false, bfr.v, (short)0, acc[mi][ni],
                false, false);
    }
}

// ---------------------------------------------------------------------------
// C[M,N] = act(A[M,K] @ W[N,K]^T + bias[N])
//   A, W bf16 row-major; 128x128 block tile, 8 waves, 32x64 per wave.
//   Double-buffered LDS, async global->LDS staging (ASYNCcnt pipelined),
//   last iteration peeled so the steady-state loop is branch-free.
//   sel (optional): device-side routing index; W += sel*wSelStride, etc.
// ---------------------------------------------------------------------------
template <typename OutT, bool RELU>
__global__ __launch_bounds__(256) void gemm_tn_wmma(
    const __bf16* __restrict__ A, const __bf16* __restrict__ W,
    const float* __restrict__ bias, const int* __restrict__ sel,
    long wSelStride, long bSelStride,
    OutT* __restrict__ C, int M, int N, int K, int ldc, int colOff)
{
    __shared__ __bf16 sA[2][128 * LDS_P];
    __shared__ __bf16 sB[2][128 * LDS_P];

    if (sel) {                      // uniform across grid; scalar path
        int s = *sel;
        W    += (long)s * wSelStride;
        bias += (long)s * bSelStride;
    }

    const int tid  = threadIdx.x;
    const int lane = tid & 31;
    const int wave = tid >> 5;
    const int wm   = wave & 3;      // wave row  (4 x 32 rows)
    const int wn   = wave >> 2;     // wave col  (2 x 64 cols)
    const int mBlock = blockIdx.y * 128;
    const int nBlock = blockIdx.x * 128;

    // staging chunk map: 512 x 16B chunks per matrix, 2 per thread
    const int r0 = tid >> 2,          c0 = (tid & 3) * 8;
    const int r1 = (tid + 256) >> 2,  c1 = c0;

    // per-thread LDS targets (byte-offset base computed once)
    __bf16* lA0[2] = { &sA[0][r0 * LDS_P + c0], &sA[1][r0 * LDS_P + c0] };
    __bf16* lA1[2] = { &sA[0][r1 * LDS_P + c1], &sA[1][r1 * LDS_P + c1] };
    __bf16* lB0[2] = { &sB[0][r0 * LDS_P + c0], &sB[1][r0 * LDS_P + c0] };
    __bf16* lB1[2] = { &sB[0][r1 * LDS_P + c1], &sB[1][r1 * LDS_P + c1] };

    // per-thread global pointers, advanced by KTILE each staged tile
    const __bf16* gA0 = A + (long)(mBlock + r0) * K + c0;
    const __bf16* gA1 = A + (long)(mBlock + r1) * K + c1;
    const __bf16* gW0 = W + (long)(nBlock + r0) * K + c0;
    const __bf16* gW1 = W + (long)(nBlock + r1) * K + c1;

    v8f acc[2][4];
#pragma unroll
    for (int mi = 0; mi < 2; ++mi)
#pragma unroll
        for (int ni = 0; ni < 4; ++ni) {
            v8f z = {0.f, 0.f, 0.f, 0.f, 0.f, 0.f, 0.f, 0.f};
            acc[mi][ni] = z;
        }

    const int laneLo = lane & 15;
    const int laneHi = lane >> 4;   // 0 or 1
    const int nk = K / KTILE;       // >= 2 for all our shapes

    // prologue: tile 0 -> buffer 0
    ASYNC_LOAD_B128(lA0[0], gA0);
    ASYNC_LOAD_B128(lA1[0], gA1);
    ASYNC_LOAD_B128(lB0[0], gW0);
    ASYNC_LOAD_B128(lB1[0], gW1);
    gA0 += KTILE; gA1 += KTILE; gW0 += KTILE; gW1 += KTILE;

    // steady state: branch-free stage(next) / wait(cur) / compute(cur)
    for (int kt = 0; kt < nk - 1; ++kt) {
        const int cur = kt & 1;
        ASYNC_LOAD_B128(lA0[cur ^ 1], gA0);
        ASYNC_LOAD_B128(lA1[cur ^ 1], gA1);
        ASYNC_LOAD_B128(lB0[cur ^ 1], gW0);
        ASYNC_LOAD_B128(lB1[cur ^ 1], gW1);
        gA0 += KTILE; gA1 += KTILE; gW0 += KTILE; gW1 += KTILE;

        WAIT_ASYNC(4);              // async loads complete in order:
        __syncthreads();            // <=4 outstanding => current tile landed
        compute_tile(sA[cur], sB[cur], acc, wm, wn, laneLo, laneHi);
        __syncthreads();
    }

    // epilogue tile (no prefetch)
    WAIT_ASYNC(0);
    __syncthreads();
    compute_tile(sA[(nk - 1) & 1], sB[(nk - 1) & 1], acc, wm, wn, laneLo, laneHi);

    // ---- epilogue: bias (+ReLU), store ----
#pragma unroll
    for (int mi = 0; mi < 2; ++mi) {
#pragma unroll
        for (int ni = 0; ni < 4; ++ni) {
            int col = nBlock + wn * 64 + ni * 16 + laneLo;
            float bv = bias[col];
#pragma unroll
            for (int r = 0; r < 8; ++r) {
                int row = mBlock + wm * 32 + mi * 16 + r + (laneHi << 3);
                float v = acc[mi][ni][r] + bv;
                if (RELU) v = fmaxf(v, 0.f);
                C[(long)row * ldc + colOff + col] = (OutT)v;
            }
        }
    }
}

// ---------------------------------------------------------------------------
extern "C" void kernel_launch(void* const* d_in, const int* in_sizes, int n_in,
                              void* d_out, int out_size, void* d_ws, size_t ws_size,
                              hipStream_t stream) {
    (void)in_sizes; (void)n_in; (void)out_size; (void)ws_size;

    const float* x      = (const float*)d_in[0];   // [B,D]
    const float* Wp     = (const float*)d_in[1];   // [T,H,D]
    const float* bp     = (const float*)d_in[2];   // [T,H]
    const float* Wop    = (const float*)d_in[3];   // [T,NOPS,H,H]
    const float* bop    = (const float*)d_in[4];   // [T,NOPS,H]
    const float* logits = (const float*)d_in[5];   // [T,NOPS]
    const float* Wr     = (const float*)d_in[6];   // [O,T*H]
    const float* br     = (const float*)d_in[7];   // [O]
    float* out = (float*)d_out;                    // [B,O]

    // ---- workspace carve-up ----
    char* w = (char*)d_ws;
    size_t off = 0;
    int*    idx  = (int*)(w + off);     off += 256;
    __bf16* xb   = (__bf16*)(w + off);  off += (size_t)BB * DD * 2;              // 16 MiB
    __bf16* Wpb  = (__bf16*)(w + off);  off += (size_t)TT * HH * DD * 2;         // 16 MiB
    __bf16* Wopb = (__bf16*)(w + off);  off += (size_t)TT * NOPS * HH * HH * 2;  // 96 MiB
    __bf16* Wrb  = (__bf16*)(w + off);  off += (size_t)OO * TT * HH * 2;         // 16 MiB
    __bf16* act0 = (__bf16*)(w + off);  off += (size_t)BB * HH * 2;              // 32 MiB
    __bf16* act1 = (__bf16*)(w + off);  off += (size_t)BB * HH * 2;              // 32 MiB
    __bf16* cat  = (__bf16*)(w + off);  off += (size_t)BB * TT * HH * 2;         // 128 MiB

    // ---- routing indices ----
    route_top2<<<1, 32, 0, stream>>>(logits, idx);

    // ---- fp32 -> bf16 conversions ----
    auto cvt = [&](const float* src, __bf16* dst, long n) {
        long threads = (n + 7) / 8;
        int blocks = (int)((threads + 255) / 256);
        cvt_f32_bf16<<<blocks, 256, 0, stream>>>(src, dst, n);
    };
    cvt(x,   xb,   (long)BB * DD);
    cvt(Wp,  Wpb,  (long)TT * HH * DD);
    cvt(Wop, Wopb, (long)TT * NOPS * HH * HH);
    cvt(Wr,  Wrb,  (long)OO * TT * HH);

    // ---- per-temper pipeline ----
    dim3 gridH(HH / 128, BB / 128);   // N=H tiles x M=B tiles
    for (int t = 0; t < TT; ++t) {
        const __bf16* WopT = Wopb + (long)t * NOPS * HH * HH;
        const float*  bopT = bop  + (long)t * NOPS * HH;

        // stage 1: act0 = relu(x @ Wp_t^T + bp_t)   (only relu(xp) is consumed)
        gemm_tn_wmma<__bf16, true><<<gridH, 256, 0, stream>>>(
            xb, Wpb + (long)t * HH * DD, bp + (long)t * HH,
            nullptr, 0, 0, act0, BB, HH, DD, HH, 0);

        // stage 2: act1 = relu(act0 @ Wa^T + ba), Wa selected by idx[t][0]
        gemm_tn_wmma<__bf16, true><<<gridH, 256, 0, stream>>>(
            act0, WopT, bopT, idx + 2 * t, (long)HH * HH, HH,
            act1, BB, HH, HH, HH, 0);

        // stage 3: cat[:, t*H:(t+1)*H] = relu(act1 @ Wb^T + bb), Wb by idx[t][1]
        gemm_tn_wmma<__bf16, true><<<gridH, 256, 0, stream>>>(
            act1, WopT, bopT, idx + 2 * t + 1, (long)HH * HH, HH,
            cat, BB, HH, HH, TT * HH, t * HH);
    }

    // ---- readout: out = cat @ Wr^T + br (fp32, no relu) ----
    dim3 gridO(OO / 128, BB / 128);
    gemm_tn_wmma<float, false><<<gridO, 256, 0, stream>>>(
        cat, Wrb, br, nullptr, 0, 0, out, BB, OO, TT * HH, OO, 0);
}
